// TransformerLayer_62122406969732
// MI455X (gfx1250) — matrix operand order repaired and checked by reading, not compiled, not run on hardware
//
#include <hip/hip_runtime.h>
#include <hip/hip_bf16.h>

#define D_MODEL 256
#define D_FF    1024
#define N_HEADS 4
#define DK      64
#define NB      4
#define NT      2048
#define M_TOK   (NB * NT)   // 8192 tokens

typedef __attribute__((ext_vector_type(16))) _Float16 v16h;
typedef __attribute__((ext_vector_type(8)))  float    v8f;

// ---------------------------------------------------------------------------
// WMMA fragment helpers (contiguous k order: K = 16*(lane/16) + elem, wave32)
// ---------------------------------------------------------------------------

__device__ __forceinline__ v8f zero8() {
  v8f z;
#pragma unroll
  for (int i = 0; i < 8; ++i) z[i] = 0.0f;
  return z;
}

// A-matrix 16x32 f16 fragment from row-major [16 x >=32] with stride lda.
// lanes 0-15: row M=lane, elems j -> K=j
// lanes 16-31: row M=lane-16, elems j -> K=16+j
__device__ __forceinline__ v16h load_frag_a(const _Float16* base, int lda) {
  const int lane = threadIdx.x & 31;
  const int m = lane & 15, hs = lane >> 4;
  const _Float16* r = base + m * lda + hs * 16;
  v16h a;
#pragma unroll
  for (int j = 0; j < 16; ++j) a[j] = r[j];
  return a;
}

// B-matrix 32x16 f16 fragment where B[k][n] = W[n][k] (W row-major, stride ldk).
// lanes 0-15: col N=lane, elems j -> K=j ; lanes 16-31: col N=lane-16, K=16+j
__device__ __forceinline__ v16h load_frag_b(const _Float16* w, int ldk) {
  const int lane = threadIdx.x & 31;
  const int n = lane & 15, hs = lane >> 4;
  const _Float16* r = w + n * ldk + hs * 16;
  v16h b;
#pragma unroll
  for (int j = 0; j < 16; ++j) b[j] = r[j];
  return b;
}

__device__ __forceinline__ v8f wmma16(v16h a, v16h b, v8f c) {
  return __builtin_amdgcn_wmma_f32_16x16x32_f16(false, a, false, b, (short)0, c,
                                                false, false);
}

// ---------------------------------------------------------------------------
// f32 -> f16 conversion
// ---------------------------------------------------------------------------
__global__ void cvt_f16_kernel(const float* __restrict__ src,
                               _Float16* __restrict__ dst, int n) {
  int i = blockIdx.x * blockDim.x + threadIdx.x;
  if (i < n) dst[i] = (_Float16)src[i];
}

// ---------------------------------------------------------------------------
// RMSNorm: one wave per row of D_MODEL=256, out f16
// ---------------------------------------------------------------------------
__global__ void rmsnorm_kernel(const float* __restrict__ x,
                               const float* __restrict__ w,
                               _Float16* __restrict__ out, int nrows) {
  const int wid = threadIdx.x >> 5;
  const int lane = threadIdx.x & 31;
  const int row = blockIdx.x * 8 + wid;
  if (row >= nrows) return;
  const float* xr = x + row * D_MODEL;
  float v[8];
  float ss = 0.0f;
#pragma unroll
  for (int j = 0; j < 8; ++j) {
    v[j] = xr[lane * 8 + j];
    ss += v[j] * v[j];
  }
#pragma unroll
  for (int m = 1; m < 32; m <<= 1) ss += __shfl_xor(ss, m, 32);
  const float inv = rsqrtf(ss * (1.0f / D_MODEL) + 1e-5f);
  _Float16* orow = out + row * D_MODEL;
#pragma unroll
  for (int j = 0; j < 8; ++j)
    orow[lane * 8 + j] = (_Float16)(v[j] * inv * w[lane * 8 + j]);
}

// ---------------------------------------------------------------------------
// GEMM: out_f16[M,N] = A[M,K] @ W[N,K]^T.  One wave -> 16 rows x 64 cols.
// ---------------------------------------------------------------------------
__global__ void gemm_f16_kernel(const _Float16* __restrict__ A,
                                const _Float16* __restrict__ W,
                                _Float16* __restrict__ out,
                                int M, int N, int K, int tn_log2) {
  const int gw = blockIdx.x * (blockDim.x >> 5) + (threadIdx.x >> 5);
  const int m0 = (gw >> tn_log2) << 4;
  const int n0 = (gw & ((1 << tn_log2) - 1)) << 6;
  if (m0 >= M) return;
  v8f acc[4];
#pragma unroll
  for (int t = 0; t < 4; ++t) acc[t] = zero8();
  for (int k0 = 0; k0 < K; k0 += 32) {
    __builtin_prefetch(A + m0 * K + k0 + 128, 0, 1);
    v16h a = load_frag_a(A + m0 * K + k0, K);
#pragma unroll
    for (int t = 0; t < 4; ++t) {
      v16h b = load_frag_b(W + (n0 + t * 16) * K + k0, K);
      acc[t] = wmma16(a, b, acc[t]);
    }
  }
  const int lane = threadIdx.x & 31;
  const int n = lane & 15, hs = lane >> 4;
#pragma unroll
  for (int t = 0; t < 4; ++t)
#pragma unroll
    for (int i = 0; i < 8; ++i)
      out[(size_t)(m0 + i + 8 * hs) * N + n0 + t * 16 + n] = (_Float16)acc[t][i];
}

// ---------------------------------------------------------------------------
// GEMM writing TRANSPOSED per-(batch,col) output:
//   vt[(b*D_MODEL + c) * NT + t]  for token m = b*NT + t, column c.
// Used for V so that attention's P@V B-fragments are contiguous loads.
// ---------------------------------------------------------------------------
__global__ void gemm_f16_vt_kernel(const _Float16* __restrict__ A,
                                   const _Float16* __restrict__ W,
                                   _Float16* __restrict__ vt,
                                   int M, int N, int K, int tn_log2) {
  const int gw = blockIdx.x * (blockDim.x >> 5) + (threadIdx.x >> 5);
  const int m0 = (gw >> tn_log2) << 4;
  const int n0 = (gw & ((1 << tn_log2) - 1)) << 6;
  if (m0 >= M) return;
  v8f acc[4];
#pragma unroll
  for (int t = 0; t < 4; ++t) acc[t] = zero8();
  for (int k0 = 0; k0 < K; k0 += 32) {
    __builtin_prefetch(A + m0 * K + k0 + 128, 0, 1);
    v16h a = load_frag_a(A + m0 * K + k0, K);
#pragma unroll
    for (int t = 0; t < 4; ++t) {
      v16h b = load_frag_b(W + (n0 + t * 16) * K + k0, K);
      acc[t] = wmma16(a, b, acc[t]);
    }
  }
  const int lane = threadIdx.x & 31;
  const int n = lane & 15, hs = lane >> 4;
#pragma unroll
  for (int t = 0; t < 4; ++t)
#pragma unroll
    for (int i = 0; i < 8; ++i) {
      const int m = m0 + i + 8 * hs;            // global token
      const int c = n0 + t * 16 + n;            // model column
      vt[((size_t)(m >> 11) * D_MODEL + c) * NT + (m & (NT - 1))] =
          (_Float16)acc[t][i];
    }
}

// ---------------------------------------------------------------------------
// GEMM + residual: out_f32[M,N] = resid[M,N] + A[M,K] @ W[N,K]^T
// ---------------------------------------------------------------------------
__global__ void gemm_resid_kernel(const _Float16* __restrict__ A,
                                  const _Float16* __restrict__ W,
                                  const float* __restrict__ resid,
                                  float* __restrict__ out,
                                  int M, int N, int K, int tn_log2) {
  const int gw = blockIdx.x * (blockDim.x >> 5) + (threadIdx.x >> 5);
  const int m0 = (gw >> tn_log2) << 4;
  const int n0 = (gw & ((1 << tn_log2) - 1)) << 6;
  if (m0 >= M) return;
  v8f acc[4];
#pragma unroll
  for (int t = 0; t < 4; ++t) acc[t] = zero8();
  for (int k0 = 0; k0 < K; k0 += 32) {
    __builtin_prefetch(A + m0 * K + k0 + 128, 0, 1);
    v16h a = load_frag_a(A + m0 * K + k0, K);
#pragma unroll
    for (int t = 0; t < 4; ++t) {
      v16h b = load_frag_b(W + (n0 + t * 16) * K + k0, K);
      acc[t] = wmma16(a, b, acc[t]);
    }
  }
  const int lane = threadIdx.x & 31;
  const int n = lane & 15, hs = lane >> 4;
#pragma unroll
  for (int t = 0; t < 4; ++t)
#pragma unroll
    for (int i = 0; i < 8; ++i) {
      const size_t idx = (size_t)(m0 + i + 8 * hs) * N + n0 + t * 16 + n;
      out[idx] = resid[idx] + acc[t][i];
    }
}

// ---------------------------------------------------------------------------
// Fused gate/up GEMM + SwiGLU: ffn_f16 = silu(A@Wg^T) * (A@Wu^T)
// ---------------------------------------------------------------------------
__global__ void gemm_swiglu_kernel(const _Float16* __restrict__ A,
                                   const _Float16* __restrict__ Wg,
                                   const _Float16* __restrict__ Wu,
                                   _Float16* __restrict__ out,
                                   int M, int N, int K, int tn_log2) {
  const int gw = blockIdx.x * (blockDim.x >> 5) + (threadIdx.x >> 5);
  const int m0 = (gw >> tn_log2) << 4;
  const int n0 = (gw & ((1 << tn_log2) - 1)) << 6;
  if (m0 >= M) return;
  v8f ag[4], au[4];
#pragma unroll
  for (int t = 0; t < 4; ++t) { ag[t] = zero8(); au[t] = zero8(); }
  for (int k0 = 0; k0 < K; k0 += 32) {
    __builtin_prefetch(A + m0 * K + k0 + 128, 0, 1);
    v16h a = load_frag_a(A + m0 * K + k0, K);
#pragma unroll
    for (int t = 0; t < 4; ++t) {
      v16h bg = load_frag_b(Wg + (n0 + t * 16) * K + k0, K);
      ag[t] = wmma16(a, bg, ag[t]);
      v16h bu = load_frag_b(Wu + (n0 + t * 16) * K + k0, K);
      au[t] = wmma16(a, bu, au[t]);
    }
  }
  const int lane = threadIdx.x & 31;
  const int n = lane & 15, hs = lane >> 4;
#pragma unroll
  for (int t = 0; t < 4; ++t)
#pragma unroll
    for (int i = 0; i < 8; ++i) {
      const float g = ag[t][i];
      const float s = g / (1.0f + __expf(-g));  // silu
      out[(size_t)(m0 + i + 8 * hs) * N + n0 + t * 16 + n] =
          (_Float16)(s * au[t][i]);
    }
}

// ---------------------------------------------------------------------------
// Flash-style causal attention.  One wave owns one 16-row Q tile of one
// (batch, head).  Block = 4 waves.  Q/K are f16 [B*T, D_MODEL] (head h in
// cols [h*64, h*64+64)); V is pre-transposed f16 vt[(b*D_MODEL + c)*T + t].
// Only the P tile round-trips LDS (C-layout -> A-layout), per-wave synced
// with s_wait_dscnt.
// ---------------------------------------------------------------------------
__global__ void attention_kernel(const _Float16* __restrict__ Q,
                                 const _Float16* __restrict__ Km,
                                 const _Float16* __restrict__ Vt,
                                 _Float16* __restrict__ O) {
  __shared__ _Float16 lds_p[4][16 * 32];   // P tile (16 q-rows x 32 keys)

  const int wid = threadIdx.x >> 5;
  const int lane = threadIdx.x & 31;
  const int n = lane & 15, hs = lane >> 4;

  const int blk_per_bh = NT / 64;                 // 32
  const int bh = blockIdx.x / blk_per_bh;         // 0..15
  const int qblk = blockIdx.x % blk_per_bh;
  const int b = bh >> 2, h = bh & 3;
  const int q0 = qblk * 64 + wid * 16;

  const _Float16* qbase = Q + (size_t)(b * NT) * D_MODEL + h * DK;
  const _Float16* kbase = Km + (size_t)(b * NT) * D_MODEL + h * DK;
  const _Float16* vtbase = Vt + (size_t)(b * D_MODEL + h * DK) * NT;

  const v16h aq0 = load_frag_a(qbase + q0 * D_MODEL + 0, D_MODEL);
  const v16h aq1 = load_frag_a(qbase + q0 * D_MODEL + 32, D_MODEL);

  float m_run[8], l_run[8];
  v8f o0 = zero8(), o1 = zero8(), o2 = zero8(), o3 = zero8();
#pragma unroll
  for (int i = 0; i < 8; ++i) { m_run[i] = -1e30f; l_run[i] = 0.0f; }

  _Float16* pls = lds_p[wid];
  const int kend = q0 + 16;  // causal: keys < kend

  for (int kb = 0; kb < kend; kb += 32) {
    // ---- S = Q K^T for 32 keys (two 16-key tiles), K-dim 64 = 2 WMMAs each
    v8f s0 = zero8(), s1 = zero8();
    s0 = wmma16(aq0, load_frag_b(kbase + kb * D_MODEL + 0, D_MODEL), s0);
    s0 = wmma16(aq1, load_frag_b(kbase + kb * D_MODEL + 32, D_MODEL), s0);
    s1 = wmma16(aq0, load_frag_b(kbase + (kb + 16) * D_MODEL + 0, D_MODEL), s1);
    s1 = wmma16(aq1, load_frag_b(kbase + (kb + 16) * D_MODEL + 32, D_MODEL), s1);

    // ---- online softmax (rows replicated across 16-lane half-groups)
    float e0[8], e1[8], alpha[8];
#pragma unroll
    for (int i = 0; i < 8; ++i) {
      const int row = q0 + i + 8 * hs;
      float v0 = s0[i] * 0.125f;            // 1/sqrt(64)
      float v1 = s1[i] * 0.125f;
      if (kb + n > row) v0 = -1e30f;        // causal mask
      if (kb + 16 + n > row) v1 = -1e30f;
      float t = fmaxf(v0, v1);
#pragma unroll
      for (int mm = 1; mm < 16; mm <<= 1) t = fmaxf(t, __shfl_xor(t, mm, 32));
      const float mn = fmaxf(m_run[i], t);
      const float al = __expf(m_run[i] - mn);
      const float x0 = __expf(v0 - mn);
      const float x1 = __expf(v1 - mn);
      float rs = x0 + x1;
#pragma unroll
      for (int mm = 1; mm < 16; mm <<= 1) rs += __shfl_xor(rs, mm, 32);
      l_run[i] = l_run[i] * al + rs;
      m_run[i] = mn;
      alpha[i] = al;
      e0[i] = x0;
      e1[i] = x1;
    }
#pragma unroll
    for (int i = 0; i < 8; ++i) {
      o0[i] *= alpha[i]; o1[i] *= alpha[i];
      o2[i] *= alpha[i]; o3[i] *= alpha[i];
    }

    // ---- P tile to LDS (C-layout -> row-major) for A-fragment reload
#pragma unroll
    for (int i = 0; i < 8; ++i) {
      const int r = i + 8 * hs;
      pls[r * 32 + n] = (_Float16)e0[i];
      pls[r * 32 + 16 + n] = (_Float16)e1[i];
    }
    asm volatile("s_wait_dscnt 0" ::: "memory");  // per-wave LDS RAW fence

    // ---- O += P @ V : B-fragments straight from pre-transposed V (global,
    //      contiguous 32B per lane), 4 d-tiles, K=32 keys
    const v16h pa = load_frag_a(pls, 32);
    o0 = wmma16(pa, load_frag_b(vtbase + 0 * 16 * NT + kb, NT), o0);
    o1 = wmma16(pa, load_frag_b(vtbase + 1 * 16 * NT + kb, NT), o1);
    o2 = wmma16(pa, load_frag_b(vtbase + 2 * 16 * NT + kb, NT), o2);
    o3 = wmma16(pa, load_frag_b(vtbase + 3 * 16 * NT + kb, NT), o3);
  }

  // ---- epilogue: normalize and store f16
  _Float16* obase = O + (size_t)(b * NT) * D_MODEL + h * DK;
#pragma unroll
  for (int i = 0; i < 8; ++i) {
    const float inv = 1.0f / l_run[i];
    const int r = q0 + i + 8 * hs;
    obase[r * D_MODEL + 0 + n] = (_Float16)(o0[i] * inv);
    obase[r * D_MODEL + 16 + n] = (_Float16)(o1[i] * inv);
    obase[r * D_MODEL + 32 + n] = (_Float16)(o2[i] * inv);
    obase[r * D_MODEL + 48 + n] = (_Float16)(o3[i] * inv);
  }
}

// ---------------------------------------------------------------------------
// Host-side orchestration
// ---------------------------------------------------------------------------
extern "C" void kernel_launch(void* const* d_in, const int* in_sizes, int n_in,
                              void* d_out, int out_size, void* d_ws, size_t ws_size,
                              hipStream_t stream) {
  (void)in_sizes; (void)n_in; (void)out_size; (void)ws_size;
  const float* x          = (const float*)d_in[0];
  // d_in[1] = causal mask (implicit in attention kernel)
  const float* rms_attn_w = (const float*)d_in[2];
  const float* wq         = (const float*)d_in[3];
  const float* wk         = (const float*)d_in[4];
  const float* wv         = (const float*)d_in[5];
  const float* wo         = (const float*)d_in[6];
  const float* rms_ffn_w  = (const float*)d_in[7];
  const float* w_gate     = (const float*)d_in[8];
  const float* w_up       = (const float*)d_in[9];
  const float* w_down     = (const float*)d_in[10];
  float* out = (float*)d_out;

  char* p = (char*)d_ws;
  auto alloc = [&](size_t bytes) -> char* {
    char* r = p;
    p += (bytes + 255) & ~(size_t)255;
    return r;
  };
  _Float16* wq_h  = (_Float16*)alloc((size_t)D_MODEL * D_MODEL * 2);
  _Float16* wk_h  = (_Float16*)alloc((size_t)D_MODEL * D_MODEL * 2);
  _Float16* wv_h  = (_Float16*)alloc((size_t)D_MODEL * D_MODEL * 2);
  _Float16* wo_h  = (_Float16*)alloc((size_t)D_MODEL * D_MODEL * 2);
  _Float16* wg_h  = (_Float16*)alloc((size_t)D_FF * D_MODEL * 2);
  _Float16* wu_h  = (_Float16*)alloc((size_t)D_FF * D_MODEL * 2);
  _Float16* wd_h  = (_Float16*)alloc((size_t)D_MODEL * D_FF * 2);
  _Float16* h16   = (_Float16*)alloc((size_t)M_TOK * D_MODEL * 2);
  _Float16* q16   = (_Float16*)alloc((size_t)M_TOK * D_MODEL * 2);
  _Float16* k16   = (_Float16*)alloc((size_t)M_TOK * D_MODEL * 2);
  _Float16* vt16  = (_Float16*)alloc((size_t)M_TOK * D_MODEL * 2);  // transposed V
  _Float16* att16 = (_Float16*)alloc((size_t)M_TOK * D_MODEL * 2);
  float*    x2    = (float*)   alloc((size_t)M_TOK * D_MODEL * 4);
  _Float16* hf16  = (_Float16*)alloc((size_t)M_TOK * D_MODEL * 2);
  _Float16* ffn16 = (_Float16*)alloc((size_t)M_TOK * D_FF * 2);

  auto cvt = [&](const float* s, _Float16* d, int n) {
    cvt_f16_kernel<<<(n + 255) / 256, 256, 0, stream>>>(s, d, n);
  };
  cvt(wq, wq_h, D_MODEL * D_MODEL);
  cvt(wk, wk_h, D_MODEL * D_MODEL);
  cvt(wv, wv_h, D_MODEL * D_MODEL);
  cvt(wo, wo_h, D_MODEL * D_MODEL);
  cvt(w_gate, wg_h, D_FF * D_MODEL);
  cvt(w_up,   wu_h, D_FF * D_MODEL);
  cvt(w_down, wd_h, D_MODEL * D_FF);

  // 1) h = rmsnorm(x)
  rmsnorm_kernel<<<M_TOK / 8, 256, 0, stream>>>(x, rms_attn_w, h16, M_TOK);

  // 2) q/k = h @ W^T (row-major f16); v -> transposed layout for attention
  {
    const int blocks = (M_TOK / 16) * (D_MODEL / 64) / 4;  // 512
    gemm_f16_kernel<<<blocks, 128, 0, stream>>>(h16, wq_h, q16, M_TOK, D_MODEL, D_MODEL, 2);
    gemm_f16_kernel<<<blocks, 128, 0, stream>>>(h16, wk_h, k16, M_TOK, D_MODEL, D_MODEL, 2);
    gemm_f16_vt_kernel<<<blocks, 128, 0, stream>>>(h16, wv_h, vt16, M_TOK, D_MODEL, D_MODEL, 2);
  }

  // 3) causal attention -> att16
  attention_kernel<<<NB * N_HEADS * (NT / 64), 128, 0, stream>>>(q16, k16, vt16, att16);

  // 4) x2 = x + att @ wo^T
  {
    const int blocks = (M_TOK / 16) * (D_MODEL / 64) / 4;
    gemm_resid_kernel<<<blocks, 128, 0, stream>>>(att16, wo_h, x, x2, M_TOK, D_MODEL, D_MODEL, 2);
  }

  // 5) hf = rmsnorm(x2)
  rmsnorm_kernel<<<M_TOK / 8, 256, 0, stream>>>(x2, rms_ffn_w, hf16, M_TOK);

  // 6) ffn = silu(hf@Wg^T) * (hf@Wu^T)
  {
    const int blocks = (M_TOK / 16) * (D_FF / 64) / 4;  // 2048
    gemm_swiglu_kernel<<<blocks, 128, 0, stream>>>(hf16, wg_h, wu_h, ffn16, M_TOK, D_FF, D_MODEL, 4);
  }

  // 7) out = x2 + ffn @ Wd^T
  {
    const int blocks = (M_TOK / 16) * (D_MODEL / 64) / 4;
    gemm_resid_kernel<<<blocks, 128, 0, stream>>>(ffn16, wd_h, x2, out, M_TOK, D_MODEL, D_FF, 2);
  }
}
